// ReactionCenterGCN_62732292326004
// MI455X (gfx1250) — matrix operand order repaired and checked by reading, not compiled, not run on hardware
//
#include <hip/hip_runtime.h>
#include <hip/hip_bf16.h>

// ---------------------------------------------------------------------------
// ReactionCenterGCN for MI455X (gfx1250, wave32).
//   1) degree + dinv = rsqrt(deg)
//   2) bufA = x @ W1                        (bf16 WMMA, f32 accum)
//   3) bufB = selfloop + edge scatter-add   (wave-per-edge, L2-resident atomics)
//   4) bufB = relu(BN1(bufB + b1))
//   5) bufA = bufB @ W2                     (bf16 WMMA, f32 accum)
//   6) bufB = selfloop + edge scatter-add
//   7) out  = sigmoid(relu(BN2(bufB+b2)) @ Wfc + bfc)
// GEMM stages LDS tiles in WMMA *fragment order* so each fragment is two
// contiguous ds_load_b128s (no v_mov_b16 repacking).
// Workspace: dinv[N] + bufA[N*128] + bufB[N*128] floats (~51.5 MB).
// ---------------------------------------------------------------------------

#define HID 128
#define BN_EPS 1e-5f

typedef __attribute__((ext_vector_type(16))) __bf16 v16bf;
typedef __attribute__((ext_vector_type(8)))  float  v8f;

__device__ __forceinline__ unsigned short f2bf(float f) {
  // round-to-nearest-even f32 -> bf16
  unsigned int u = __float_as_uint(f);
  u += 0x7FFFu + ((u >> 16) & 1u);
  return (unsigned short)(u >> 16);
}

// ---------------- degree / dinv ----------------
__global__ void k_fill1(float* __restrict__ p, int n) {
  int i = blockIdx.x * blockDim.x + threadIdx.x;
  if (i < n) p[i] = 1.0f;  // self-loop contributes 1 to every degree
}

__global__ void k_deg(const int* __restrict__ dst, float* __restrict__ deg, int E) {
  int i = blockIdx.x * blockDim.x + threadIdx.x;
  if (i < E) atomicAdd(&deg[dst[i]], 1.0f);
}

__global__ void k_rsqrt_inplace(float* __restrict__ p, int n) {
  int i = blockIdx.x * blockDim.x + threadIdx.x;
  if (i < n) {
    float d = p[i];
    p[i] = (d > 0.0f) ? rsqrtf(d) : 0.0f;
  }
}

// ---------------- GEMM: Y[M,128] = X[M,K] @ W[K,128], bf16 WMMA f32-acc ------
// Block = 256 threads (8 waves). Each block handles MT 16-row M-tiles; wave w
// owns output columns [16w, 16w+16). KP in {32,128} = K padded to 32.
//
// LDS is staged in fragment order:
//  A frag slot (ISA 7.12.2, 16-bit A 16x32): lane = m + 16*((kc>>3)&1),
//                                            i    = (kc&7) + 8*(kc>>4)
//  B frag slot (32x16): lane = (n&15) + 16*((kc>>3)&1), i = (kc&7) + 8*(kc>>4), wave = n>>4
// so every fragment = one aligned 32-byte per-lane LDS read.
template <int KP, int MT>
__global__ __launch_bounds__(256)
void k_gemm_bf16(const float* __restrict__ X, const float* __restrict__ W,
                 float* __restrict__ Y, int nrows, int K) {
  constexpr int CH = KP / 32;                        // 32-wide K chunks
  __shared__ __align__(32) unsigned short sA[MT * CH * 32 * 16];
  __shared__ __align__(32) unsigned short sB[CH * 8 * 32 * 16];

  const int tid   = threadIdx.x;
  const int mBase = blockIdx.x * (MT * 16);

  // Stage W (KP x 128) -> fragment-ordered bf16, zero-padded rows K..KP.
  for (int idx = tid; idx < KP * HID; idx += 256) {
    int k = idx / HID, n = idx % HID;
    unsigned short v = (k < K) ? f2bf(W[k * HID + n]) : (unsigned short)0;
    int ch = k >> 5, kc = k & 31;
    int lane = (n & 15) + 16 * ((kc >> 3) & 1);
    int i    = (kc & 7) + 8 * (kc >> 4);
    int wv   = n >> 4;
    sB[(((ch * 8 + wv) * 32) + lane) * 16 + i] = v;
  }
  // Stage X tile (MT*16 x KP) -> fragment-ordered bf16 (rows clamped).
  for (int idx = tid; idx < MT * 16 * KP; idx += 256) {
    int r = idx / KP, k = idx % KP;
    int row = mBase + r; if (row >= nrows) row = nrows - 1;
    unsigned short v = (k < K) ? f2bf(X[(size_t)row * K + k]) : (unsigned short)0;
    int mt = r >> 4, m = r & 15;
    int ch = k >> 5, kc = k & 31;
    int lane = m + 16 * ((kc >> 3) & 1);
    int i    = (kc & 7) + 8 * (kc >> 4);
    sA[(((mt * CH + ch) * 32) + lane) * 16 + i] = v;
  }
  __syncthreads();

  const int wave = tid >> 5;
  const int lane = tid & 31;
  const int half = lane >> 4;
  const int l15  = lane & 15;
  const int nBase = wave * 16;

  // Hoist this wave's B fragments (CH x v16bf) into registers.
  v16bf bfrag[CH];
#pragma unroll
  for (int ch = 0; ch < CH; ++ch)
    bfrag[ch] = *reinterpret_cast<const v16bf*>(&sB[(((ch * 8 + wave) * 32) + lane) * 16]);

#pragma unroll
  for (int mt = 0; mt < MT; ++mt) {
    v8f c = {};
#pragma unroll
    for (int ch = 0; ch < CH; ++ch) {
      v16bf a = *reinterpret_cast<const v16bf*>(&sA[(((mt * CH + ch) * 32) + lane) * 16]);
      c = __builtin_amdgcn_wmma_f32_16x16x32_bf16(
          /*neg_a=*/false, a, /*neg_b=*/false, bfrag[ch],
          /*c_mod=*/(short)0, c, /*reuse_a=*/false, /*reuse_b=*/false);
    }
    // C/D: VGPR j -> row j (lanes 0-15) / j+8 (lanes 16-31), col = lane&15.
    const int rBase = mBase + mt * 16;
    float* yp = Y + (size_t)(rBase + half * 8) * HID + nBase + l15;
    if (rBase + 16 <= nrows) {          // uniform fast path: unguarded stores
#pragma unroll
      for (int j = 0; j < 8; ++j) yp[(size_t)j * HID] = c[j];
    } else {                            // ragged tail
#pragma unroll
      for (int j = 0; j < 8; ++j) {
        int row = rBase + j + half * 8;
        if (row < nrows) Y[(size_t)row * HID + nBase + l15] = c[j];
      }
    }
  }
}

// ---------------- self-loop init: out[i,:] = h[i,:] * dinv[i]^2 --------------
__global__ void k_selfloop(const float* __restrict__ h, const float* __restrict__ dinv,
                           float* __restrict__ out, int N) {
  int i = blockIdx.x * blockDim.x + threadIdx.x;   // one thread = one float4
  if (i >= N * 32) return;
  int node = i >> 5, q = i & 31;
  float dd = dinv[node]; dd *= dd;
  float4 v = ((const float4*)h)[(size_t)node * 32 + q];
  float4 r; r.x = v.x * dd; r.y = v.y * dd; r.z = v.z * dd; r.w = v.w * dd;
  ((float4*)out)[(size_t)node * 32 + q] = r;
}

// ---------------- wave-per-edge scatter-add ----------------------------------
__global__ __launch_bounds__(256)
void k_scatter(const int* __restrict__ src, const int* __restrict__ dst,
               const float* __restrict__ dinv, const float* __restrict__ h,
               float* __restrict__ out, int E) {
  int wid  = (blockIdx.x * blockDim.x + threadIdx.x) >> 5;
  int lane = threadIdx.x & 31;
  if (wid >= E) return;
  int s = src[wid], d = dst[wid];
  float norm = dinv[s] * dinv[d];
  float4 v = ((const float4*)(h + (size_t)s * HID))[lane];
  float* o = out + (size_t)d * HID + lane * 4;
  atomicAdd(o + 0, v.x * norm);
  atomicAdd(o + 1, v.y * norm);
  atomicAdd(o + 2, v.z * norm);
  atomicAdd(o + 3, v.w * norm);
}

// ---------------- fused bias + BN + ReLU (in place) --------------------------
__global__ void k_bnrelu(float* __restrict__ h, const float* __restrict__ bias,
                         const float* __restrict__ gamma, const float* __restrict__ beta,
                         const float* __restrict__ mean, const float* __restrict__ var,
                         int total) {
  int i = blockIdx.x * blockDim.x + threadIdx.x;
  if (i >= total) return;
  int f = i & (HID - 1);
  float t = (h[i] + bias[f] - mean[f]) * rsqrtf(var[f] + BN_EPS) * gamma[f] + beta[f];
  h[i] = fmaxf(t, 0.0f);
}

// ---------------- head: BN2+ReLU+FC+sigmoid, one wave per node ---------------
__global__ __launch_bounds__(256)
void k_head(const float* __restrict__ h, const float* __restrict__ bias,
            const float* __restrict__ gamma, const float* __restrict__ beta,
            const float* __restrict__ mean, const float* __restrict__ var,
            const float* __restrict__ Wfc, const float* __restrict__ bfc,
            float* __restrict__ out, int N) {
  int wid  = (blockIdx.x * blockDim.x + threadIdx.x) >> 5;
  int lane = threadIdx.x & 31;
  if (wid >= N) return;
  float s = 0.0f;
#pragma unroll
  for (int j = 0; j < 4; ++j) {
    int f = lane + 32 * j;
    float t = (h[(size_t)wid * HID + f] + bias[f] - mean[f]) *
                  rsqrtf(var[f] + BN_EPS) * gamma[f] + beta[f];
    t = fmaxf(t, 0.0f);
    s += t * Wfc[f];
  }
#pragma unroll
  for (int off = 16; off > 0; off >>= 1) s += __shfl_xor(s, off, 32);
  if (lane == 0) out[wid] = 1.0f / (1.0f + __expf(-(s + bfc[0])));
}

// ---------------------------------------------------------------------------
extern "C" void kernel_launch(void* const* d_in, const int* in_sizes, int n_in,
                              void* d_out, int out_size, void* d_ws, size_t ws_size,
                              hipStream_t stream) {
  const float* x      = (const float*)d_in[0];
  const int*   edge   = (const int*)  d_in[1];  // [2,E]: row0=src, row1=dst
  const float* W1     = (const float*)d_in[2];
  const float* b1     = (const float*)d_in[3];
  const float* gamma1 = (const float*)d_in[4];
  const float* beta1  = (const float*)d_in[5];
  const float* mean1  = (const float*)d_in[6];
  const float* var1   = (const float*)d_in[7];
  const float* W2     = (const float*)d_in[8];
  const float* b2     = (const float*)d_in[9];
  const float* gamma2 = (const float*)d_in[10];
  const float* beta2  = (const float*)d_in[11];
  const float* mean2  = (const float*)d_in[12];
  const float* var2   = (const float*)d_in[13];
  const float* Wfc    = (const float*)d_in[14];
  const float* bfc    = (const float*)d_in[15];
  float* out = (float*)d_out;

  const int Fin = in_sizes[2] / HID;          // W1 is [Fin, 128]
  const int N   = in_sizes[0] / Fin;
  const int E   = in_sizes[1] / 2;
  const int* src = edge;
  const int* dst = edge + E;

  float* dinv = (float*)d_ws;                 // N
  float* bufA = dinv + N;                     // N*128
  float* bufB = bufA + (size_t)N * HID;       // N*128

  const int T = 256;
  constexpr int MT = 4;                       // 64 rows per GEMM block
  const int gRows = MT * 16;
  const int gemmGrid = (N + gRows - 1) / gRows;

  // 1) degrees -> dinv
  k_fill1<<<(N + T - 1) / T, T, 0, stream>>>(dinv, N);
  k_deg<<<(E + T - 1) / T, T, 0, stream>>>(dst, dinv, E);
  k_rsqrt_inplace<<<(N + T - 1) / T, T, 0, stream>>>(dinv, N);

  // 2) bufA = x @ W1   (Fin=22 -> KP=32)
  if (Fin <= 32) {
    k_gemm_bf16<32, MT><<<gemmGrid, T, 0, stream>>>(x, W1, bufA, N, Fin);
  } else {
    k_gemm_bf16<128, MT><<<gemmGrid, T, 0, stream>>>(x, W1, bufA, N, Fin);
  }

  // 3) aggregate layer 1 into bufB
  k_selfloop<<<(N * 32 + T - 1) / T, T, 0, stream>>>(bufA, dinv, bufB, N);
  k_scatter<<<(int)(((size_t)E * 32 + T - 1) / T), T, 0, stream>>>(src, dst, dinv, bufA, bufB, E);

  // 4) bufB = relu(BN1(bufB + b1))
  k_bnrelu<<<(int)(((size_t)N * HID + T - 1) / T), T, 0, stream>>>(bufB, b1, gamma1, beta1, mean1, var1, N * HID);

  // 5) bufA = bufB @ W2   (K=128)
  k_gemm_bf16<128, MT><<<gemmGrid, T, 0, stream>>>(bufB, W2, bufA, N, HID);

  // 6) aggregate layer 2 into bufB
  k_selfloop<<<(N * 32 + T - 1) / T, T, 0, stream>>>(bufA, dinv, bufB, N);
  k_scatter<<<(int)(((size_t)E * 32 + T - 1) / T), T, 0, stream>>>(src, dst, dinv, bufA, bufB, E);

  // 7) head: BN2 + ReLU + FC + sigmoid
  k_head<<<(int)(((size_t)N * 32 + T - 1) / T), T, 0, stream>>>(bufB, b2, gamma2, beta2, mean2, var2, Wfc, bfc, out, N);
}
